// TAN_61014305407751
// MI455X (gfx1250) — matrix operand order repaired and checked by reading, not compiled, not run on hardware
//
#include <hip/hip_runtime.h>
#include <hip/hip_bf16.h>

// Bidirectional masked LSTM + attention head for MI455X (gfx1250, wave32).
//   1) weights packed once into WMMA-fragment-linear bf16 layout
//   2) big WMMA GEMM: G[L][B][4H] = Xp @ W_ih^T + b   (both dirs)
//   3) 128 sequential steps: G[t] += h_p @ W_hh^T (WMMA accumulate) ; cell
//   4) small VALU kernels for attention softmax / context / MLP head
// Operands are stored k-chunk packed so every WMMA fragment is loaded with
// b128 global loads (no LDS gathers).  Workspace: ~464 MiB.

typedef __bf16 bf;
typedef __attribute__((ext_vector_type(16))) __bf16 v16bf;
typedef __attribute__((ext_vector_type(8)))  float  v8f;

#define BB   256
#define LL   128
#define DD   1024
#define HH   512
#define G4H  2048
#define LINN 512

#define BM 128   // 8 waves x 16 rows
#define BN 64    // 4 x 16-col WMMA tiles per wave
#define BK 32

union FragA { uint4 q[2]; v16bf v; };

__device__ __forceinline__ float sigm(float x) { return 1.0f / (1.0f + __expf(-x)); }

// ---------------------------------------------------------------------------
// Packed layouts (kt = k / 32, kk = k % 32):
//   Ap[(kt*M + m)*32 + kk]   (M x K activation matrix)
//   Bp[(kt*N + n)*32 + kk]   (K x N weight matrix, i.e. W^T pre-packed)
// C[M][N] fp32 row-major:  C = Ap*Bp (+bias[col] | +=C).
// M%BM==0, N%BN==0, K%BK==0 at every call site.
// ---------------------------------------------------------------------------
__global__ __launch_bounds__(256) void gemm_bf16(const bf* __restrict__ Ap,
                                                 const bf* __restrict__ Bp,
                                                 float* __restrict__ C,
                                                 int M, int N, int K,
                                                 const float* __restrict__ bias,
                                                 int accum)
{
    const int tid  = threadIdx.x;
    const int wave = tid >> 5;
    const int lane = tid & 31;
    const int m0   = blockIdx.y * BM;
    const int n0   = blockIdx.x * BN;

    const v8f zero = {0.f, 0.f, 0.f, 0.f, 0.f, 0.f, 0.f, 0.f};
    v8f acc[4] = {zero, zero, zero, zero};

    const int half = lane >> 4;     // K sub-range selector
    const int l16  = lane & 15;     // M (A) / N (B) index within tile
    const int am   = m0 + wave * 16 + l16;
    const int kh   = half * 8;      // A: elems 0-7 -> K=kh..kh+7, 8-15 -> K=16+kh..
    const int kb   = half * 16;     // unused: B loads use kh, same k order as A

    const int KT = K >> 5;
    for (int kt = 0; kt < KT; ++kt) {
        // A fragment: two contiguous 16B chunks
        const bf* ab = Ap + ((size_t)kt * M + am) * 32;
        FragA af;
        af.q[0] = *(const uint4*)(ab + kh);
        af.q[1] = *(const uint4*)(ab + 16 + kh);
        if (kt + 1 < KT)
            __builtin_prefetch(ab + (size_t)M * 32, 0, 1);

#pragma unroll
        for (int nt = 0; nt < 4; ++nt) {
            const int nn = n0 + nt * 16 + l16;
            const bf* bb = Bp + ((size_t)kt * N + nn) * 32 + kh;
            FragA bfg;
            bfg.q[0] = *(const uint4*)(bb);
            bfg.q[1] = *(const uint4*)(bb + 16);
            acc[nt] = __builtin_amdgcn_wmma_f32_16x16x32_bf16(
                false, af.v, false, bfg.v, (short)0, acc[nt], false, false);
        }
    }

    // C/D layout: VGPR r -> M = r + 8*half, N = lane&15
    const int cm = m0 + wave * 16 + half * 8;
#pragma unroll
    for (int nt = 0; nt < 4; ++nt) {
        const int cn = n0 + nt * 16 + l16;
        const float bv = bias ? bias[cn] : 0.f;
#pragma unroll
        for (int r = 0; r < 8; ++r) {
            float v = acc[nt][r] + bv;
            float* p = C + (size_t)(cm + r) * N + cn;
            if (accum) v += *p;
            *p = v;
        }
    }
}

// ---------------------------------------------------------------------------
// Weight pack: Bp[(kt*R + n)*32 + kk] = bf16(W[n][kt*32 + kk]);  R = 4H = 2048.
// ---------------------------------------------------------------------------
__global__ void k_w_pack(const float* __restrict__ W, bf* __restrict__ Bp, int Kd)
{
    int o = blockIdx.x * blockDim.x + threadIdx.x;
    if (o >= Kd * G4H) return;
    int kk   = o & 31;
    int rest = o >> 5;
    int n    = rest & (G4H - 1);
    int kt   = rest >> 11;
    Bp[o] = (bf)W[(size_t)n * Kd + kt * 32 + kk];
}

// Ap pack of x, rows m = t*B + b (M = L*B = 32768), K = D.
__global__ void k_x_fwd(const float* __restrict__ x, bf* __restrict__ Ap)
{
    int o = blockIdx.x * blockDim.x + threadIdx.x;
    if (o >= LL * BB * DD) return;
    int kk   = o & 31;
    int rest = o >> 5;
    int m    = rest & (LL * BB - 1);
    int kt   = rest >> 15;
    int b    = m & (BB - 1);
    int t    = m >> 8;
    Ap[o] = (bf)x[((size_t)b * LL + t) * DD + kt * 32 + kk];
}

// Same, with valid-length time reversal: row (t,b) <- x[b][len[b]-1-t] or 0.
__global__ void k_x_rev(const float* __restrict__ x, const int* __restrict__ len,
                        bf* __restrict__ Ap)
{
    int o = blockIdx.x * blockDim.x + threadIdx.x;
    if (o >= LL * BB * DD) return;
    int kk   = o & 31;
    int rest = o >> 5;
    int m    = rest & (LL * BB - 1);
    int kt   = rest >> 15;
    int b    = m & (BB - 1);
    int t    = m >> 8;
    int lb   = len[b];
    float v = 0.f;
    if (t < lb) v = x[((size_t)b * LL + (lb - 1 - t)) * DD + kt * 32 + kk];
    Ap[o] = (bf)v;
}

// ---------------------------------------------------------------------------
// LSTM pointwise cell.  S = G + t*B*4H (gate order i|f|g|o).
// Writes h in packed-A layout for the next step's recurrent GEMM.
// ---------------------------------------------------------------------------
__global__ void k_cell(const float* __restrict__ S, const int* __restrict__ len,
                       int t, float* __restrict__ h, float* __restrict__ c,
                       bf* __restrict__ hp, float* __restrict__ outslice)
{
    int idx = blockIdx.x * blockDim.x + threadIdx.x;  // B*H
    int j = idx & (HH - 1);
    int b = idx >> 9;
    const float* Sb = S + (size_t)b * G4H;
    float ig = sigm(Sb[j]);
    float fg = sigm(Sb[HH + j]);
    float gg = tanhf(Sb[2 * HH + j]);
    float og = sigm(Sb[3 * HH + j]);
    float cn = fg * c[idx] + ig * gg;
    float hn = og * tanhf(cn);
    bool valid = t < len[b];
    float h2 = valid ? hn : h[idx];
    float c2 = valid ? cn : c[idx];
    h[idx] = h2;
    c[idx] = c2;
    hp[((size_t)(j >> 5) * BB + b) * 32 + (j & 31)] = (bf)h2;  // packed A
    outslice[idx] = valid ? hn : 0.f;
}

// tw[b][d] = mean_k target_word[b][0][k][d]
__global__ void k_tw(const float* __restrict__ twin, float* __restrict__ tw)
{
    int idx = blockIdx.x * blockDim.x + threadIdx.x;  // B*D
    int d = idx & (DD - 1);
    int b = idx >> 10;
    float s = 0.f;
    for (int k = 0; k < 5; ++k) s += twin[((size_t)b * 5 + k) * DD + d];
    tw[idx] = 0.2f * s;
}

// u[b][l] = x[b][l]·Wh[:D] + tw[b]·Wh[D:] + b_tanh[l] (masked) -> softmax row
__global__ __launch_bounds__(256) void k_attn(const float* __restrict__ x,
                                              const float* __restrict__ tw,
                                              const float* __restrict__ Wh,
                                              const float* __restrict__ btanh,
                                              const int* __restrict__ len,
                                              float* __restrict__ alphas)
{
    const int b = blockIdx.x;
    const int tid = threadIdx.x;
    const int wave = tid >> 5, lane = tid & 31;
    __shared__ float red[256];
    __shared__ float us[LL];

    float p = 0.f;
    for (int d = tid; d < DD; d += 256) p += tw[b * DD + d] * Wh[DD + d];
    red[tid] = p; __syncthreads();
    for (int s = 128; s > 0; s >>= 1) { if (tid < s) red[tid] += red[tid + s]; __syncthreads(); }
    const float s2 = red[0];
    __syncthreads();

    const int lb = len[b];
    for (int l = wave; l < LL; l += 8) {
        float sum = 0.f;
        const float* xr = x + ((size_t)b * LL + l) * DD;
        for (int d = lane; d < DD; d += 32) sum += xr[d] * Wh[d];
        for (int off = 16; off; off >>= 1) sum += __shfl_down(sum, off, 32);
        if (lane == 0) us[l] = (l < lb) ? (sum + s2 + btanh[l]) : -1.0e6f;
    }
    __syncthreads();

    float m = (tid < LL) ? us[tid] : -3.0e38f;
    red[tid] = m; __syncthreads();
    for (int s = 128; s > 0; s >>= 1) { if (tid < s) red[tid] = fmaxf(red[tid], red[tid + s]); __syncthreads(); }
    m = red[0]; __syncthreads();
    float e = (tid < LL) ? __expf(us[tid] - m) : 0.f;
    red[tid] = e; __syncthreads();
    for (int s = 128; s > 0; s >>= 1) { if (tid < s) red[tid] += red[tid + s]; __syncthreads(); }
    float inv = 1.f / red[0];
    if (tid < LL) alphas[b * LL + tid] = e * inv;
}

// ctx[b][j] = sum_l alpha[b][l] * hcat[b][l][j]; backward half un-reversed here.
__global__ void k_context(const float* __restrict__ outf,     // [L][B][H]
                          const float* __restrict__ outbrev,  // [L][B][H]
                          const float* __restrict__ alphas,   // [B][L]
                          const int* __restrict__ len,
                          float* __restrict__ ctx)            // [B][2H]
{
    int idx = blockIdx.x * blockDim.x + threadIdx.x;  // B*2H
    int j = idx & (2 * HH - 1);
    int b = idx >> 10;
    int lb = len[b];
    float s = 0.f;
    if (j < HH) {
        for (int l = 0; l < LL; ++l)
            s += alphas[b * LL + l] * outf[((size_t)l * BB + b) * HH + j];
    } else {
        int jb = j - HH;
        for (int l = 0; l < lb; ++l)
            s += alphas[b * LL + l] * outbrev[((size_t)(lb - 1 - l) * BB + b) * HH + jb];
    }
    ctx[idx] = s;
}

__global__ void k_lin(const float* __restrict__ ctx, const float* __restrict__ Wl,
                      const float* __restrict__ bl, float* __restrict__ lin)
{
    int idx = blockIdx.x * blockDim.x + threadIdx.x;  // B*LIN
    int n = idx & (LINN - 1);
    int b = idx >> 9;
    const float* cr = ctx + (size_t)b * 2 * HH;
    const float* wr = Wl + (size_t)n * 2 * HH;
    float s = bl[n];
    for (int d = 0; d < 2 * HH; ++d) s += cr[d] * wr[d];
    lin[idx] = fmaxf(s, 0.f);
}

__global__ void k_out(const float* __restrict__ lin, const float* __restrict__ Wo,
                      const float* __restrict__ bo, float* __restrict__ out)
{
    int idx = blockIdx.x * blockDim.x + threadIdx.x;  // B*3
    if (idx >= BB * 3) return;
    int k = idx % 3;
    int b = idx / 3;
    float s = bo[k];
    for (int d = 0; d < LINN; ++d) s += lin[b * LINN + d] * Wo[k * LINN + d];
    out[idx] = s;
}

// ---------------------------------------------------------------------------
extern "C" void kernel_launch(void* const* d_in, const int* in_sizes, int n_in,
                              void* d_out, int out_size, void* d_ws, size_t ws_size,
                              hipStream_t stream)
{
    (void)in_sizes; (void)n_in; (void)out_size; (void)ws_size;
    const float* x     = (const float*)d_in[0];
    const int*   xlen  = (const int*)  d_in[1];
    const float* twin  = (const float*)d_in[3];
    const float* Wihf  = (const float*)d_in[5];
    const float* Whhf  = (const float*)d_in[6];
    const float* bfv   = (const float*)d_in[7];
    const float* Wihb  = (const float*)d_in[8];
    const float* Whhb  = (const float*)d_in[9];
    const float* bbv   = (const float*)d_in[10];
    const float* Wh    = (const float*)d_in[11];
    const float* btanh = (const float*)d_in[12];
    const float* Wlin  = (const float*)d_in[13];
    const float* blin  = (const float*)d_in[14];
    const float* Wout  = (const float*)d_in[15];
    const float* bout  = (const float*)d_in[16];

    char* ws = (char*)d_ws;
    size_t off = 0;
    auto alloc = [&](size_t bytes) -> char* {
        char* p = ws + off;
        off += (bytes + 255) & ~(size_t)255;
        return p;
    };
    bf*    wihf_p  = (bf*)   alloc((size_t)DD * G4H * 2);
    bf*    wihb_p  = (bf*)   alloc((size_t)DD * G4H * 2);
    bf*    whhf_p  = (bf*)   alloc((size_t)HH * G4H * 2);
    bf*    whhb_p  = (bf*)   alloc((size_t)HH * G4H * 2);
    bf*    xp      = (bf*)   alloc((size_t)LL * BB * DD * 2);
    float* G       = (float*)alloc((size_t)LL * BB * G4H * 4);
    float* outf    = (float*)alloc((size_t)LL * BB * HH * 4);
    float* outbrev = (float*)alloc((size_t)LL * BB * HH * 4);
    float* hbuf    = (float*)alloc((size_t)BB * HH * 4);
    float* cbuf    = (float*)alloc((size_t)BB * HH * 4);
    bf*    hp      = (bf*)   alloc((size_t)BB * HH * 2);
    float* tw      = (float*)alloc((size_t)BB * DD * 4);
    float* alphas  = (float*)alloc((size_t)BB * LL * 4);
    float* ctx     = (float*)alloc((size_t)BB * 2 * HH * 4);
    float* lin     = (float*)alloc((size_t)BB * LINN * 4);

    // 1) weight packing (fragment-linear bf16)
    {
        int n1 = DD * G4H;
        k_w_pack<<<(n1 + 255) / 256, 256, 0, stream>>>(Wihf, wihf_p, DD);
        k_w_pack<<<(n1 + 255) / 256, 256, 0, stream>>>(Wihb, wihb_p, DD);
        int n2 = HH * G4H;
        k_w_pack<<<(n2 + 255) / 256, 256, 0, stream>>>(Whhf, whhf_p, HH);
        k_w_pack<<<(n2 + 255) / 256, 256, 0, stream>>>(Whhb, whhb_p, HH);
    }

    const int  nx = LL * BB * DD;
    const dim3 ggBig(G4H / BN, (LL * BB) / BM);  // 32 x 256 blocks
    const dim3 ggStep(G4H / BN, BB / BM);        // 32 x 2 blocks

    // 2) forward direction
    k_x_fwd<<<nx / 256, 256, 0, stream>>>(x, xp);
    gemm_bf16<<<ggBig, 256, 0, stream>>>(xp, wihf_p, G, LL * BB, G4H, DD, bfv, 0);
    hipMemsetAsync(hbuf, 0, (size_t)BB * HH * 4, stream);
    hipMemsetAsync(cbuf, 0, (size_t)BB * HH * 4, stream);
    hipMemsetAsync(hp,   0, (size_t)BB * HH * 2, stream);
    for (int t = 0; t < LL; ++t) {
        float* St = G + (size_t)t * BB * G4H;
        gemm_bf16<<<ggStep, 256, 0, stream>>>(hp, whhf_p, St, BB, G4H, HH, nullptr, 1);
        k_cell<<<(BB * HH) / 256, 256, 0, stream>>>(St, xlen, t, hbuf, cbuf, hp,
                                                    outf + (size_t)t * BB * HH);
    }

    // 3) backward direction (time-reversed valid sequence)
    k_x_rev<<<nx / 256, 256, 0, stream>>>(x, xlen, xp);
    gemm_bf16<<<ggBig, 256, 0, stream>>>(xp, wihb_p, G, LL * BB, G4H, DD, bbv, 0);
    hipMemsetAsync(hbuf, 0, (size_t)BB * HH * 4, stream);
    hipMemsetAsync(cbuf, 0, (size_t)BB * HH * 4, stream);
    hipMemsetAsync(hp,   0, (size_t)BB * HH * 2, stream);
    for (int t = 0; t < LL; ++t) {
        float* St = G + (size_t)t * BB * G4H;
        gemm_bf16<<<ggStep, 256, 0, stream>>>(hp, whhb_p, St, BB, G4H, HH, nullptr, 1);
        k_cell<<<(BB * HH) / 256, 256, 0, stream>>>(St, xlen, t, hbuf, cbuf, hp,
                                                    outbrev + (size_t)t * BB * HH);
    }

    // 4) attention + head
    k_tw<<<(BB * DD) / 256, 256, 0, stream>>>(twin, tw);
    k_attn<<<BB, 256, 0, stream>>>(x, tw, Wh, btanh, xlen, alphas);
    k_context<<<(BB * 2 * HH) / 256, 256, 0, stream>>>(outf, outbrev, alphas, xlen, ctx);
    k_lin<<<(BB * LINN) / 256, 256, 0, stream>>>(ctx, Wlin, blin, lin);
    k_out<<<(BB * 3 + 255) / 256, 256, 0, stream>>>(lin, Wout, bout, (float*)d_out);
}
